// Zorro_AVT_53781580481000
// MI455X (gfx1250) — matrix operand order repaired and checked by reading, not compiled, not run on hardware
//
#include <hip/hip_runtime.h>
#include <hip/hip_bf16.h>
#include <cmath>

// ---------------- types for WMMA ----------------
typedef __attribute__((ext_vector_type(16))) __bf16 bf16x16;
typedef __attribute__((ext_vector_type(8)))  float  f32x8;

union FragBF { bf16x16 v; uint4 q[2]; };

// ---------------- problem constants ----------------
constexpr int BZ = 4, NSEG_V = 512, NSEG_A = 256, NSEG_T = 128, NFUS = 8;
constexpr int NTOK   = NSEG_V + NSEG_A + NSEG_T + NFUS;   // 904
constexpr int DIMC   = 512;
constexpr int DEPTHC = 6;
constexpr int HEADSC = 8;
constexpr int DHC    = 64;
constexpr int MROWS  = BZ * NTOK;                         // 3616 (== 113 * 32)
constexpr int INNERC = 1365;
constexpr int FF1PAD = 2752;   // >= 2*INNER (2730), multiple of 64
constexpr int FF2K   = 1376;   // >= INNER, multiple of 32
constexpr int OUTDC  = 768;
constexpr float EPSC = 1e-5f;

// ---------------- helpers ----------------
__device__ __forceinline__ f32x8 zero8() {
  f32x8 z;
#pragma unroll
  for (int i = 0; i < 8; ++i) z[i] = 0.0f;
  return z;
}

__device__ __forceinline__ f32x8 wmma_bf16(const FragBF& a, const FragBF& b, f32x8 c) {
  return __builtin_amdgcn_wmma_f32_16x16x32_bf16(false, a.v, false, b.v, (short)0, c, false, false);
}

// block reduce over 256 threads (8 waves, wave32); sbuf must hold >= 9 floats
__device__ __forceinline__ float blockReduceSum256(float v, float* sbuf) {
  int lane = threadIdx.x & 31;
  int w = threadIdx.x >> 5;
#pragma unroll
  for (int o = 16; o > 0; o >>= 1) v += __shfl_down(v, o, 32);
  if (lane == 0) sbuf[w] = v;
  __syncthreads();
  if (threadIdx.x == 0) {
    float s = 0.f;
    for (int i = 0; i < 8; ++i) s += sbuf[i];
    sbuf[8] = s;
  }
  __syncthreads();
  float r = sbuf[8];
  __syncthreads();
  return r;
}

// ---------------- concat tokens: [video|audio|text|fusion] ----------------
__global__ void concat_kernel(const float* __restrict__ vd, const float* __restrict__ au,
                              const float* __restrict__ tx, const float* __restrict__ fu,
                              float* __restrict__ T) {
  long idx = (long)blockIdx.x * blockDim.x + threadIdx.x;
  long tot = (long)MROWS * DIMC;
  if (idx >= tot) return;
  int d = (int)(idx % DIMC);
  long m = idx / DIMC;
  int b = (int)(m / NTOK);
  int p = (int)(m % NTOK);
  float v;
  if (p < NSEG_V)                   v = vd[((long)b * NSEG_V + p) * DIMC + d];
  else if (p < NSEG_V + NSEG_A)     v = au[((long)b * NSEG_A + (p - NSEG_V)) * DIMC + d];
  else if (p < NSEG_V + NSEG_A + NSEG_T)
                                    v = tx[((long)b * NSEG_T + (p - NSEG_V - NSEG_A)) * DIMC + d];
  else                              v = fu[(long)(p - NSEG_V - NSEG_A - NSEG_T) * DIMC + d];
  T[idx] = v;
}

// ---------------- per-key pad mask (fusion keys always masked) ----------------
__global__ void keymask_kernel(const unsigned char* __restrict__ vm,
                               const unsigned char* __restrict__ am,
                               const unsigned char* __restrict__ tm,
                               unsigned char* __restrict__ KM) {
  int idx = blockIdx.x * blockDim.x + threadIdx.x;
  if (idx >= BZ * NTOK) return;
  int b = idx / NTOK, p = idx % NTOK;
  unsigned char m;
  if (p < NSEG_V)              m = vm[b * NSEG_V + p];
  else if (p < NSEG_V + NSEG_A) m = am[b * NSEG_A + (p - NSEG_V)];
  else if (p < NSEG_V + NSEG_A + NSEG_T)
                               m = tm[b * NSEG_T + (p - NSEG_V - NSEG_A)];
  else                         m = 1;   // fusion keys: pad mask is all-True (faithful)
  KM[idx] = m ? 1 : 0;
}

// ---------------- weight convert fp32 [K][Ncol] -> bf16 transposed [Ncpad][Kpad], zero pad ----------------
__global__ void cvtw_kernel(const float* __restrict__ src, __bf16* __restrict__ dst,
                            int K, int Ncol, int Kpad, int Ncpad,
                            long srcStride, long dstStride) {
  long l = blockIdx.y;
  long idx = (long)blockIdx.x * blockDim.x + threadIdx.x;
  long tot = (long)Ncpad * Kpad;
  if (idx >= tot) return;
  int n = (int)(idx / Kpad);
  int k = (int)(idx % Kpad);
  float v = (n < Ncol && k < K) ? src[l * srcStride + (long)k * Ncol + n] : 0.0f;
  dst[l * dstStride + idx] = (__bf16)v;
}

// ---------------- layernorm fp32 row -> bf16 row (one 256-thread block per row) ----------------
__global__ void ln_kernel(const float* __restrict__ X, const float* __restrict__ gamma,
                          __bf16* __restrict__ Y) {
  __shared__ float sbuf[9];
  long row = blockIdx.x;
  int t = threadIdx.x;
  const float* x = X + row * DIMC;
  float x0 = x[t], x1 = x[t + 256];
  float mu = blockReduceSum256(x0 + x1, sbuf) * (1.0f / DIMC);
  float d0 = x0 - mu, d1 = x1 - mu;
  float var = blockReduceSum256(d0 * d0 + d1 * d1, sbuf) * (1.0f / DIMC);
  float inv = rsqrtf(var + EPSC);
  __bf16* y = Y + row * DIMC;
  y[t]       = (__bf16)(d0 * inv * gamma[t]);
  y[t + 256] = (__bf16)(d1 * inv * gamma[t + 256]);
}

// ---------------- WMMA GEMM: C[M x Nc] = A[M x K](bf16) * Wt^T, Wt stored [Nc][K] bf16 ----------------
// One wave computes a 32x64 tile (2 m-subtiles x 4 n-subtiles = 8 accumulators); each B fragment
// is reused by two back-to-back WMMAs to double arithmetic intensity out of L2.
// MODE 0: f32 out, 1: f32 out + residual, 2: bf16 out * scale
template <int MODE>
__global__ void gemm_kernel(const __bf16* __restrict__ A, const __bf16* __restrict__ Wt,
                            const float* __restrict__ Res, void* __restrict__ Out,
                            int K, int nTiles64, int ldc, float scale, int mTiles32) {
  int wid = (int)(((long)blockIdx.x * blockDim.x + threadIdx.x) >> 5);
  if (wid >= mTiles32 * nTiles64) return;
  int mt = wid / nTiles64;
  int nt = wid % nTiles64;
  int lane = threadIdx.x & 31;
  int r = lane & 15;
  int hh = lane >> 4;
  long m0 = (long)mt * 32;
  long n0 = (long)nt * 64;

  f32x8 acc[2][4];
#pragma unroll
  for (int s = 0; s < 2; ++s)
#pragma unroll
    for (int j = 0; j < 4; ++j) acc[s][j] = zero8();

  const __bf16* arow0 = A + (m0 + r) * (long)K;
  const __bf16* arow1 = A + (m0 + 16 + r) * (long)K;
  for (int k0 = 0; k0 < K; k0 += 32) {
    FragBF af[2];
    af[0].q[0] = *(const uint4*)(arow0 + k0 + hh * 16);
    af[0].q[1] = *(const uint4*)(arow0 + k0 + hh * 16 + 8);
    af[1].q[0] = *(const uint4*)(arow1 + k0 + hh * 16);
    af[1].q[1] = *(const uint4*)(arow1 + k0 + hh * 16 + 8);
    if (k0 + 32 < K) {
      __builtin_prefetch(arow0 + k0 + 32, 0, 1);   // global_prefetch_b8
      __builtin_prefetch(arow1 + k0 + 32, 0, 1);
    }
#pragma unroll
    for (int j = 0; j < 4; ++j) {
      FragBF bf_;
      const __bf16* bp = Wt + (n0 + j * 16 + r) * (long)K + k0 + hh * 16;
      bf_.q[0] = *(const uint4*)(bp);
      bf_.q[1] = *(const uint4*)(bp + 8);
      acc[0][j] = wmma_bf16(af[0], bf_, acc[0][j]);   // same B operand back-to-back
      acc[1][j] = wmma_bf16(af[1], bf_, acc[1][j]);
    }
  }

#pragma unroll
  for (int s = 0; s < 2; ++s) {
#pragma unroll
    for (int j = 0; j < 4; ++j) {
      long ncol = n0 + j * 16 + r;
#pragma unroll
      for (int v = 0; v < 8; ++v) {
        long row = m0 + s * 16 + v + hh * 8;
        long off = row * (long)ldc + ncol;
        float val = acc[s][j][v];
        if (MODE == 0)      ((float*)Out)[off] = val;
        else if (MODE == 1) ((float*)Out)[off] = val + Res[off];
        else                ((__bf16*)Out)[off] = (__bf16)(val * scale);
      }
    }
  }
}

// ---------------- transpose V out of kv buffer: VT[b][h][d][n] ----------------
__global__ void vtrans_kernel(const __bf16* __restrict__ KV, __bf16* __restrict__ VT) {
  long idx = (long)blockIdx.x * blockDim.x + threadIdx.x;
  long tot = (long)BZ * HEADSC * DHC * NTOK;
  if (idx >= tot) return;
  int n = (int)(idx % NTOK);
  long rem = idx / NTOK;
  int d = (int)(rem % DHC); rem /= DHC;
  int h = (int)(rem % HEADSC);
  int b = (int)(rem / HEADSC);
  VT[idx] = KV[((long)b * NTOK + n) * (2 * HEADSC * DHC) + (HEADSC * DHC) + h * DHC + d];
}

// ---------------- block-diagonal flash attention (two-pass), one wave per (b,h,16-query tile) ----------------
__global__ void attn_kernel(const __bf16* __restrict__ Q, const __bf16* __restrict__ KV,
                            const __bf16* __restrict__ VT, const unsigned char* __restrict__ KM,
                            __bf16* __restrict__ O) {
  constexpr int TPB = 57;  // 32 video + 16 audio + 8 text + 1 fusion tile per batch
  __shared__ __align__(16) __bf16 ptile[4][16][32];
  __shared__ float mrow[4][16];
  __shared__ float srow[4][16];

  int wid = (int)(((long)blockIdx.x * blockDim.x + threadIdx.x) >> 5);
  if (wid >= BZ * HEADSC * TPB) return;
  int w = (threadIdx.x >> 5) & 3;
  int t = wid % TPB;
  int rem = wid / TPB;
  int h = rem % HEADSC;
  int b = rem / HEADSC;
  int lane = threadIdx.x & 31;
  int r = lane & 15;
  int hh = lane >> 4;

  int ks, ke; bool fus = false;
  if (t < 32)      { ks = 0;   ke = 512; }
  else if (t < 48) { ks = 512; ke = 768; }
  else if (t < 56) { ks = 768; ke = 896; }
  else             { ks = 0;   ke = 896; fus = true; }

  int q0 = t * 16;
  long mbase = (long)b * NTOK;
  int qr = (fus && r >= 8) ? 7 : r;   // clamp loads for partial fusion tile

  // Q fragments (16x64 split into two 16x32 A-frags), already scaled by 1/8 at GEMM time
  const __bf16* qrow = Q + (mbase + q0 + qr) * (long)DIMC + h * DHC;
  FragBF aq[2];
#pragma unroll
  for (int c = 0; c < 2; ++c) {
    aq[c].q[0] = *(const uint4*)(qrow + c * 32 + hh * 16);
    aq[c].q[1] = *(const uint4*)(qrow + c * 32 + hh * 16 + 8);
  }

  float rm[8], rs[8];
#pragma unroll
  for (int v = 0; v < 8; ++v) { rm[v] = -3.0e38f; rs[v] = 0.0f; }

  // ---- pass 1: running softmax stats over key tiles of 16 ----
  for (int k0 = ks; k0 < ke; k0 += 16) {
    int key = k0 + r;   // this lane's key column
    const __bf16* krow = KV + (mbase + key) * (long)(2 * HEADSC * DHC) + h * DHC;
    f32x8 sim = zero8();
#pragma unroll
    for (int c = 0; c < 2; ++c) {
      FragBF bk;
      bk.q[0] = *(const uint4*)(krow + c * 32 + hh * 16);
      bk.q[1] = *(const uint4*)(krow + c * 32 + hh * 16 + 8);
      sim = wmma_bf16(aq[c], bk, sim);
    }
    bool masked = KM[mbase + key] != 0;
#pragma unroll
    for (int v = 0; v < 8; ++v) {
      float x = masked ? -3.0e38f : sim[v];
      float mx = x;
      mx = fmaxf(mx, __shfl_xor(mx, 1, 32));
      mx = fmaxf(mx, __shfl_xor(mx, 2, 32));
      mx = fmaxf(mx, __shfl_xor(mx, 4, 32));
      mx = fmaxf(mx, __shfl_xor(mx, 8, 32));
      float mnew = fmaxf(rm[v], mx);
      float e = __expf(x - mnew);
      float se = e;
      se += __shfl_xor(se, 1, 32);
      se += __shfl_xor(se, 2, 32);
      se += __shfl_xor(se, 4, 32);
      se += __shfl_xor(se, 8, 32);
      rs[v] = rs[v] * __expf(rm[v] - mnew) + se;
      rm[v] = mnew;
    }
  }

  if (r == 0) {
#pragma unroll
    for (int v = 0; v < 8; ++v) {
      mrow[w][hh * 8 + v] = rm[v];
      srow[w][hh * 8 + v] = rs[v] > 0.0f ? 1.0f / rs[v] : 0.0f;
    }
  }
  asm volatile("s_wait_dscnt 0x0" ::: "memory");
  float mv[8], iv[8];
#pragma unroll
  for (int v = 0; v < 8; ++v) { mv[v] = mrow[w][hh * 8 + v]; iv[v] = srow[w][hh * 8 + v]; }

  // ---- pass 2: P = exp(sim-m)/s  ->  O += P(16x32) @ V(32x64) ----
  f32x8 oacc[4];
#pragma unroll
  for (int j = 0; j < 4; ++j) oacc[j] = zero8();
  long vtbase = ((long)(b * HEADSC + h) * DHC) * NTOK;

  for (int k0 = ks; k0 < ke; k0 += 32) {
#pragma unroll
    for (int cc = 0; cc < 2; ++cc) {
      int key = k0 + cc * 16 + r;
      const __bf16* krow = KV + (mbase + key) * (long)(2 * HEADSC * DHC) + h * DHC;
      f32x8 sim = zero8();
#pragma unroll
      for (int c = 0; c < 2; ++c) {
        FragBF bk;
        bk.q[0] = *(const uint4*)(krow + c * 32 + hh * 16);
        bk.q[1] = *(const uint4*)(krow + c * 32 + hh * 16 + 8);
        sim = wmma_bf16(aq[c], bk, sim);
      }
      bool masked = KM[mbase + key] != 0;
#pragma unroll
      for (int v = 0; v < 8; ++v) {
        float pe = masked ? 0.0f : __expf(sim[v] - mv[v]) * iv[v];
        ptile[w][hh * 8 + v][cc * 16 + r] = (__bf16)pe;   // element (row, keycol)
      }
    }
    asm volatile("s_wait_dscnt 0x0" ::: "memory");
    FragBF ap;
    ap.q[0] = *(const uint4*)(&ptile[w][r][hh * 16]);
    ap.q[1] = *(const uint4*)(&ptile[w][r][hh * 16 + 8]);
#pragma unroll
    for (int j = 0; j < 4; ++j) {
      FragBF bv;
      const __bf16* vp = VT + vtbase + (long)(j * 16 + r) * NTOK + k0 + hh * 16;
      bv.q[0] = *(const uint4*)vp;
      bv.q[1] = *(const uint4*)(vp + 8);
      oacc[j] = wmma_bf16(ap, bv, oacc[j]);
    }
    asm volatile("s_wait_dscnt 0x0" ::: "memory");
  }

  // store O (guard invalid rows of the partial fusion tile)
  if (!(fus && hh)) {
#pragma unroll
    for (int j = 0; j < 4; ++j) {
#pragma unroll
      for (int v = 0; v < 8; ++v) {
        O[(mbase + q0 + v + hh * 8) * (long)DIMC + h * DHC + j * 16 + r] = (__bf16)oacc[j][v];
      }
    }
  }
}

// ---------------- GEGLU: h = gelu(gate) * a, write K-padded bf16 ----------------
__global__ void geglu_kernel(const __bf16* __restrict__ FF, __bf16* __restrict__ H) {
  long idx = (long)blockIdx.x * blockDim.x + threadIdx.x;
  long tot = (long)MROWS * FF2K;
  if (idx >= tot) return;
  int i = (int)(idx % FF2K);
  long m = idx / FF2K;
  float out = 0.0f;
  if (i < INNERC) {
    float a = (float)FF[m * FF1PAD + i];
    float g = (float)FF[m * FF1PAD + INNERC + i];
    float gl = 0.5f * g * (1.0f + erff(g * 0.70710678118654752f));  // exact gelu
    out = gl * a;
  }
  H[idx] = (__bf16)out;
}

// ---------------- final LN + fusion-token mean pool ----------------
__global__ void lnpool_kernel(const float* __restrict__ T, const float* __restrict__ gamma,
                              float* __restrict__ pooled) {
  __shared__ float sbuf[9];
  int b = blockIdx.x;
  int t = threadIdx.x;
  float a0 = 0.0f, a1 = 0.0f;
  for (int f = 0; f < NFUS; ++f) {
    const float* x = T + ((long)b * NTOK + (NTOK - NFUS) + f) * DIMC;
    float x0 = x[t], x1 = x[t + 256];
    float mu = blockReduceSum256(x0 + x1, sbuf) * (1.0f / DIMC);
    float d0 = x0 - mu, d1 = x1 - mu;
    float var = blockReduceSum256(d0 * d0 + d1 * d1, sbuf) * (1.0f / DIMC);
    float inv = rsqrtf(var + EPSC);
    a0 += d0 * inv * gamma[t];
    a1 += d1 * inv * gamma[t + 256];
  }
  pooled[(long)b * DIMC + t]       = a0 * (1.0f / NFUS);
  pooled[(long)b * DIMC + t + 256] = a1 * (1.0f / NFUS);
}

// ---------------- output projection (tiny: 4x768) ----------------
__global__ void outproj_kernel(const float* __restrict__ pooled, const float* __restrict__ Wout,
                               float* __restrict__ out) {
  int idx = blockIdx.x * blockDim.x + threadIdx.x;
  if (idx >= BZ * OUTDC) return;
  int b = idx / OUTDC, o = idx % OUTDC;
  float acc = 0.0f;
  for (int d = 0; d < DIMC; ++d) acc += pooled[b * DIMC + d] * Wout[(long)d * OUTDC + o];
  out[idx] = acc;
}

// ---------------- host orchestration ----------------
extern "C" void kernel_launch(void* const* d_in, const int* in_sizes, int n_in,
                              void* d_out, int out_size, void* d_ws, size_t ws_size,
                              hipStream_t stream) {
  (void)in_sizes; (void)n_in; (void)out_size; (void)ws_size;
  const float* video = (const float*)d_in[0];
  const float* audio = (const float*)d_in[1];
  const float* text  = (const float*)d_in[2];
  const unsigned char* vmask = (const unsigned char*)d_in[3];
  const unsigned char* amask = (const unsigned char*)d_in[4];
  const unsigned char* tmask = (const unsigned char*)d_in[5];
  const float* fusion = (const float*)d_in[6];
  const float* g_attn = (const float*)d_in[7];
  const float* Wq     = (const float*)d_in[8];
  const float* Wkv    = (const float*)d_in[9];
  const float* Wo     = (const float*)d_in[10];
  const float* g_ff   = (const float*)d_in[11];
  const float* Wff1   = (const float*)d_in[12];
  const float* Wff2   = (const float*)d_in[13];
  const float* g_fin  = (const float*)d_in[14];
  const float* Wout   = (const float*)d_in[15];
  float* outp = (float*)d_out;

  // workspace carve-out
  char* p = (char*)d_ws;
  auto alloc = [&](size_t bytes) -> char* {
    char* r = p;
    p += (bytes + 255) & ~(size_t)255;
    return r;
  };
  float*  tokens = (float*)alloc((size_t)MROWS * DIMC * 4);
  __bf16* xb     = (__bf16*)alloc((size_t)MROWS * DIMC * 2);       // LN output (reused for both LNs)
  __bf16* qb     = (__bf16*)alloc((size_t)MROWS * DIMC * 2);
  __bf16* kvb    = (__bf16*)alloc((size_t)MROWS * 2 * DIMC * 2);
  __bf16* vt     = (__bf16*)alloc((size_t)BZ * HEADSC * DHC * NTOK * 2);
  __bf16* ob     = (__bf16*)alloc((size_t)MROWS * DIMC * 2);
  __bf16* ffb    = (__bf16*)alloc((size_t)MROWS * FF1PAD * 2);
  __bf16* hb     = (__bf16*)alloc((size_t)MROWS * FF2K * 2);
  unsigned char* km = (unsigned char*)alloc((size_t)BZ * NTOK);
  float*  pooled = (float*)alloc((size_t)BZ * DIMC * 4);
  __bf16* wqt    = (__bf16*)alloc((size_t)DEPTHC * DIMC * DIMC * 2);
  __bf16* wkvt   = (__bf16*)alloc((size_t)DEPTHC * 2 * DIMC * DIMC * 2);
  __bf16* wot    = (__bf16*)alloc((size_t)DEPTHC * DIMC * DIMC * 2);
  __bf16* wff1t  = (__bf16*)alloc((size_t)DEPTHC * FF1PAD * DIMC * 2);
  __bf16* wff2t  = (__bf16*)alloc((size_t)DEPTHC * DIMC * FF2K * 2);

  // ---- one-time per launch: weight convert (transposed, padded) ----
  {
    long tot;
    tot = (long)DIMC * DIMC;
    cvtw_kernel<<<dim3((unsigned)((tot + 255) / 256), DEPTHC), 256, 0, stream>>>(
        Wq, wqt, DIMC, DIMC, DIMC, DIMC, (long)DIMC * DIMC, (long)DIMC * DIMC);
    tot = (long)(2 * DIMC) * DIMC;
    cvtw_kernel<<<dim3((unsigned)((tot + 255) / 256), DEPTHC), 256, 0, stream>>>(
        Wkv, wkvt, DIMC, 2 * DIMC, DIMC, 2 * DIMC, (long)DIMC * 2 * DIMC, (long)2 * DIMC * DIMC);
    tot = (long)DIMC * DIMC;
    cvtw_kernel<<<dim3((unsigned)((tot + 255) / 256), DEPTHC), 256, 0, stream>>>(
        Wo, wot, DIMC, DIMC, DIMC, DIMC, (long)DIMC * DIMC, (long)DIMC * DIMC);
    tot = (long)FF1PAD * DIMC;
    cvtw_kernel<<<dim3((unsigned)((tot + 255) / 256), DEPTHC), 256, 0, stream>>>(
        Wff1, wff1t, DIMC, 2 * INNERC, DIMC, FF1PAD, (long)DIMC * 2 * INNERC, (long)FF1PAD * DIMC);
    tot = (long)DIMC * FF2K;
    cvtw_kernel<<<dim3((unsigned)((tot + 255) / 256), DEPTHC), 256, 0, stream>>>(
        Wff2, wff2t, INNERC, DIMC, FF2K, DIMC, (long)INNERC * DIMC, (long)DIMC * FF2K);
  }

  // ---- concat tokens + key mask ----
  {
    long tot = (long)MROWS * DIMC;
    concat_kernel<<<(unsigned)((tot + 255) / 256), 256, 0, stream>>>(video, audio, text, fusion, tokens);
    keymask_kernel<<<(BZ * NTOK + 255) / 256, 256, 0, stream>>>(vmask, amask, tmask, km);
  }

  const int mT = MROWS / 32;  // 113 row tiles of 32
  auto gemm_blocks = [&](int nT) { return (unsigned)((mT * nT * 32 + 127) / 128); };

  for (int l = 0; l < DEPTHC; ++l) {
    const __bf16* wq_l   = wqt   + (size_t)l * DIMC * DIMC;
    const __bf16* wkv_l  = wkvt  + (size_t)l * 2 * DIMC * DIMC;
    const __bf16* wo_l   = wot   + (size_t)l * DIMC * DIMC;
    const __bf16* wff1_l = wff1t + (size_t)l * FF1PAD * DIMC;
    const __bf16* wff2_l = wff2t + (size_t)l * DIMC * FF2K;

    // x = LN(tokens, g_attn[l])  (bf16)
    ln_kernel<<<MROWS, 256, 0, stream>>>(tokens, g_attn + (size_t)l * DIMC, xb);

    // q = (x @ Wq) * DH^-0.5  (bf16),  kv = x @ Wkv  (bf16)
    gemm_kernel<2><<<gemm_blocks(DIMC / 64), 128, 0, stream>>>(
        xb, wq_l, nullptr, qb, DIMC, DIMC / 64, DIMC, 0.125f, mT);
    gemm_kernel<2><<<gemm_blocks(2 * DIMC / 64), 128, 0, stream>>>(
        xb, wkv_l, nullptr, kvb, DIMC, 2 * DIMC / 64, 2 * DIMC, 1.0f, mT);

    // transpose V for WMMA B-fragment-friendly layout
    {
      long tot = (long)BZ * HEADSC * DHC * NTOK;
      vtrans_kernel<<<(unsigned)((tot + 255) / 256), 256, 0, stream>>>(kvb, vt);
    }

    // block-diagonal flash attention -> ob (bf16)
    {
      int waves = BZ * HEADSC * 57;
      attn_kernel<<<(unsigned)((waves * 32 + 127) / 128), 128, 0, stream>>>(qb, kvb, vt, km, ob);
    }

    // tokens = ob @ Wo + tokens  (in-place residual)
    gemm_kernel<1><<<gemm_blocks(DIMC / 64), 128, 0, stream>>>(
        ob, wo_l, tokens, tokens, DIMC, DIMC / 64, DIMC, 1.0f, mT);

    // y = LN(tokens, g_ff[l])  (bf16, reuse xb)
    ln_kernel<<<MROWS, 256, 0, stream>>>(tokens, g_ff + (size_t)l * DIMC, xb);

    // ff = y @ Wff1  (bf16, N padded to 2752)
    gemm_kernel<2><<<gemm_blocks(FF1PAD / 64), 128, 0, stream>>>(
        xb, wff1_l, nullptr, ffb, DIMC, FF1PAD / 64, FF1PAD, 1.0f, mT);

    // h = gelu(gate) * a  (bf16, K padded to 1376 with zeros)
    {
      long tot = (long)MROWS * FF2K;
      geglu_kernel<<<(unsigned)((tot + 255) / 256), 256, 0, stream>>>(ffb, hb);
    }

    // tokens = h @ Wff2 + tokens  (in-place residual, K = 1376)
    gemm_kernel<1><<<gemm_blocks(DIMC / 64), 128, 0, stream>>>(
        hb, wff2_l, tokens, tokens, FF2K, DIMC / 64, DIMC, 1.0f, mT);
  }

  // final LN + mean pool over fusion tokens, then output projection
  lnpool_kernel<<<BZ, 256, 0, stream>>>(tokens, g_fin, pooled);
  outproj_kernel<<<(BZ * OUTDC + 255) / 256, 256, 0, stream>>>(pooled, Wout, outp);
}
